// HierarchicalGeometricAttentionMLP_49263274885651
// MI455X (gfx1250) — compile-verified, matrix operand order checked by reading, not run on hardware
//
#include <hip/hip_runtime.h>
#include <math.h>
#include <float.h>

#define E       128
#define HHEADS  8
#define DHEAD   16
#define KNBR    32
#define TOPK_MAXN 2048

typedef float     v2f  __attribute__((ext_vector_type(2)));
typedef float     v8f  __attribute__((ext_vector_type(8)));
typedef _Float16  v16h __attribute__((ext_vector_type(16)));

// ---------------------------------------------------------------------------
// WMMA 16x16 tile GEMM:  C[16x16] = A(16xK) * Bt(16xK)^T  (+bias, +sanitize)
//   A  : row-major, lda floats per row, rows m0..m0+15 (A already offset)
//   Bt : row-major, ldb floats per row, rows n0..n0+15 (in_w layout: out[n][m] = sum_k A[n][k]*Bt[m][k])
//   C  : row-major, ldc floats per row (already offset to (m0,n0))
// One wave32 per tile. EXEC all-ones (full wave, no divergence).
// ---------------------------------------------------------------------------
#if __has_builtin(__builtin_amdgcn_wmma_f32_16x16x4_f32)
__device__ __forceinline__ void gemm_tile(const float* __restrict__ A, int lda,
                                          const float* __restrict__ Bt, int ldb,
                                          float* __restrict__ C, int ldc,
                                          const float* __restrict__ bias,
                                          int Kdim, bool do_sanitize)
{
    const int lane = threadIdx.x & 31;
    const int lo   = lane & 15;   // A: M row / B: N col / C: N col
    const int hi   = lane >> 4;   // K-half select (and C: M-half select)
    v8f acc = {0.f, 0.f, 0.f, 0.f, 0.f, 0.f, 0.f, 0.f};
    const float* arow = A  + (size_t)lo * lda + 2 * hi;
    const float* brow = Bt + (size_t)lo * ldb + 2 * hi;
    // V_WMMA_F32_16X16X4_F32: A frag = A[lo][kk + 2*hi + v], B frag = Bt[lo][kk + 2*hi + v]
    for (int kk = 0; kk < Kdim; kk += 4) {
        v2f a; a.x = arow[kk]; a.y = arow[kk + 1];
        v2f b; b.x = brow[kk]; b.y = brow[kk + 1];
        acc = __builtin_amdgcn_wmma_f32_16x16x4_f32(false, a, false, b, (short)0, acc, false, false);
    }
    const float bn = bias ? bias[lo] : 0.0f;
    #pragma unroll
    for (int r = 0; r < 8; ++r) {              // C VGPR r -> row (r + 8*hi), col lo
        float v = acc[r] + bn;
        if (do_sanitize) {
            v = (v != v) ? 0.0f : v;           // nan_to_num
            v = fminf(fmaxf(v, -1.0e4f), 1.0e4f);
        }
        C[(size_t)(r + 8 * hi) * ldc + lo] = v;
    }
}
#else
// Fallback: f16 WMMA (codegen-verified builtin). Fragment layout per ISA 7.12.2:
// 16-bit A 16x32: lanes0-15 VGPR0..3 -> K=0..7, VGPR4..7 -> K=16..23; lanes16-31 +8.
__device__ __forceinline__ void gemm_tile(const float* __restrict__ A, int lda,
                                          const float* __restrict__ Bt, int ldb,
                                          float* __restrict__ C, int ldc,
                                          const float* __restrict__ bias,
                                          int Kdim, bool do_sanitize)
{
    const int lane = threadIdx.x & 31;
    const int lo   = lane & 15;
    const int hi   = lane >> 4;
    v8f acc = {0.f, 0.f, 0.f, 0.f, 0.f, 0.f, 0.f, 0.f};
    for (int kk = 0; kk < Kdim; kk += 32) {
        v16h a, b;
        #pragma unroll
        for (int e = 0; e < 16; ++e) {
            int k = kk + ((e >> 3) << 4) + (hi << 3) + (e & 7);
            a[e] = (_Float16)A [(size_t)lo * lda + k];
            b[e] = (_Float16)Bt[(size_t)lo * ldb + k];
        }
        acc = __builtin_amdgcn_wmma_f32_16x16x32_f16(false, a, false, b, (short)0, acc, false, false);
    }
    const float bn = bias ? bias[lo] : 0.0f;
    #pragma unroll
    for (int r = 0; r < 8; ++r) {
        float v = acc[r] + bn;
        if (do_sanitize) {
            v = (v != v) ? 0.0f : v;
            v = fminf(fmaxf(v, -1.0e4f), 1.0e4f);
        }
        C[(size_t)(r + 8 * hi) * ldc + lo] = v;
    }
}
#endif

// ---------------------------------------------------------------------------
// K1: centroids, positional encoder (Linear3->32, exact GELU, LayerNorm32),
//     pos_enhanced = [x[:, :96] + prompt[:96] , pe].  One wave32 per row.
// ---------------------------------------------------------------------------
__global__ void prep_kernel(const float* __restrict__ x, const float* __restrict__ coords9,
                            const float* __restrict__ prompt,
                            const float* __restrict__ pe_w, const float* __restrict__ pe_b,
                            const float* __restrict__ pe_g, const float* __restrict__ pe_bt,
                            float* __restrict__ pos, float* __restrict__ cent, int n)
{
    const int i    = blockIdx.x;
    const int lane = threadIdx.x;           // 0..31
    if (i >= n) return;
    const float* c9 = coords9 + (size_t)i * 9;
    const float cx = (c9[0] + c9[3] + c9[6]) * (1.0f / 3.0f);
    const float cy = (c9[1] + c9[4] + c9[7]) * (1.0f / 3.0f);
    const float cz = (c9[2] + c9[5] + c9[8]) * (1.0f / 3.0f);
    if (lane == 0) { cent[i * 4 + 0] = cx; cent[i * 4 + 1] = cy; cent[i * 4 + 2] = cz; }

    // Linear(3,32) + exact GELU
    float z = cx * pe_w[lane] + cy * pe_w[32 + lane] + cz * pe_w[64 + lane] + pe_b[lane];
    float h = 0.5f * z * (1.0f + erff(z * 0.70710678118654752f));
    // LayerNorm over 32 lanes (wave32 shuffle reductions)
    float s = h;
    for (int off = 16; off > 0; off >>= 1) s += __shfl_xor(s, off, 32);
    const float mu = s * (1.0f / 32.0f);
    float d = h - mu;
    float vs = d * d;
    for (int off = 16; off > 0; off >>= 1) vs += __shfl_xor(vs, off, 32);
    const float var = vs * (1.0f / 32.0f);
    const float pe  = d * rsqrtf(var + 1e-5f) * pe_g[lane] + pe_bt[lane];

    float* prow = pos + (size_t)i * E;
    const float* xrow = x + (size_t)i * E;
    #pragma unroll
    for (int r = 0; r < 3; ++r) {           // cols 0..95 : x + prompt
        int c = lane + 32 * r;
        prow[c] = xrow[c] + prompt[c];
    }
    prow[96 + lane] = pe;                   // cols 96..127 : pe
}

// ---------------------------------------------------------------------------
// K2: top-K (K=32 smallest d^2, lowest-index tie-break == lax.top_k on -dist).
//     One 256-thread block per query; d^2 array in LDS; 32x block argmin.
// ---------------------------------------------------------------------------
__global__ void topk_kernel(const float* __restrict__ cent, int* __restrict__ nbr, int n)
{
    __shared__ float sd[TOPK_MAXN];
    __shared__ float rv[256];
    __shared__ int   ri[256];
    const int i   = blockIdx.x;
    const int tid = threadIdx.x;
    const float cx = cent[i * 4 + 0], cy = cent[i * 4 + 1], cz = cent[i * 4 + 2];
    for (int j = tid; j < n; j += 256) {
        const float dx = cx - cent[j * 4 + 0];
        const float dy = cy - cent[j * 4 + 1];
        const float dz = cz - cent[j * 4 + 2];
        sd[j] = dx * dx + dy * dy + dz * dz;
    }
    __syncthreads();
    for (int k = 0; k < KNBR; ++k) {
        float bv = FLT_MAX; int bi = 0x7FFFFFFF;
        for (int j = tid; j < n; j += 256) {
            const float v = sd[j];
            if (v < bv || (v == bv && j < bi)) { bv = v; bi = j; }
        }
        rv[tid] = bv; ri[tid] = bi;
        __syncthreads();
        for (int stp = 128; stp > 0; stp >>= 1) {
            if (tid < stp) {
                const float ov = rv[tid + stp]; const int oi = ri[tid + stp];
                if (ov < rv[tid] || (ov == rv[tid] && oi < ri[tid])) { rv[tid] = ov; ri[tid] = oi; }
            }
            __syncthreads();
        }
        if (tid == 0) { nbr[(size_t)i * KNBR + k] = ri[0]; sd[ri[0]] = FLT_MAX; }
        __syncthreads();
    }
}

// ---------------------------------------------------------------------------
// K3: qkv = pos(N,128) @ in_w(384,128)^T + in_b   -- WMMA, one wave per tile
// ---------------------------------------------------------------------------
__global__ void qkv_kernel(const float* __restrict__ pos, const float* __restrict__ in_w,
                           const float* __restrict__ in_b, float* __restrict__ qkv)
{
    const int nt = blockIdx.x;   // 0..23
    const int mt = blockIdx.y;   // 0..N/16-1
    gemm_tile(pos + (size_t)mt * 16 * E, E,
              in_w + (size_t)nt * 16 * E, E,
              qkv + (size_t)mt * 16 * (3 * E) + nt * 16, 3 * E,
              in_b + nt * 16, E, false);
}

// ---------------------------------------------------------------------------
// K4: sparse attention over top-K neighbors (exactly equals masked softmax:
//     exp(-1e9 + ...) underflows to 0 in fp32).  One block/query, wave/head.
// ---------------------------------------------------------------------------
__global__ void attn_kernel(const float* __restrict__ qkv, const int* __restrict__ nbr,
                            float* __restrict__ o)
{
    __shared__ int   s_nbr[KNBR];
    __shared__ float s_p[HHEADS][KNBR];
    const int i    = blockIdx.x;
    const int tid  = threadIdx.x;
    const int h    = tid >> 5;
    const int lane = tid & 31;
    if (tid < KNBR) s_nbr[tid] = nbr[(size_t)i * KNBR + tid];
    __syncthreads();

    const int j = s_nbr[lane];
    const float* qrow = qkv + (size_t)i * 384 + h * DHEAD;
    const float* krow = qkv + (size_t)j * 384 + 128 + h * DHEAD;
    float s = 0.0f;
    #pragma unroll
    for (int d = 0; d < DHEAD; ++d) s += qrow[d] * krow[d];
    s *= 0.25f;                                   // 1/sqrt(16)
    float m = s;
    for (int off = 16; off > 0; off >>= 1) m = fmaxf(m, __shfl_xor(m, off, 32));
    const float e = expf(s - m);
    float es = e;
    for (int off = 16; off > 0; off >>= 1) es += __shfl_xor(es, off, 32);
    s_p[h][lane] = e / es;
    __syncthreads();

    if (lane < DHEAD) {
        float acc = 0.0f;
        for (int l = 0; l < KNBR; ++l) {
            const int jj = s_nbr[l];
            acc += s_p[h][l] * qkv[(size_t)jj * 384 + 256 + h * DHEAD + lane];
        }
        o[(size_t)i * E + h * DHEAD + lane] = acc;
    }
}

// ---------------------------------------------------------------------------
// K5: out = o(N,128) @ out_w(128,128)^T + out_b, fused sanitize -- WMMA
// ---------------------------------------------------------------------------
__global__ void outproj_kernel(const float* __restrict__ ov, const float* __restrict__ out_w,
                               const float* __restrict__ out_b, float* __restrict__ out)
{
    const int nt = blockIdx.x;   // 0..7
    const int mt = blockIdx.y;   // 0..N/16-1
    gemm_tile(ov + (size_t)mt * 16 * E, E,
              out_w + (size_t)nt * 16 * E, E,
              out + (size_t)mt * 16 * E + nt * 16, E,
              out_b + nt * 16, E, true);
}

// ---------------------------------------------------------------------------
extern "C" void kernel_launch(void* const* d_in, const int* in_sizes, int n_in,
                              void* d_out, int out_size, void* d_ws, size_t ws_size,
                              hipStream_t stream)
{
    const float* x      = (const float*)d_in[0];
    const float* coords = (const float*)d_in[1];
    const float* prompt = (const float*)d_in[2];
    const float* pe_w   = (const float*)d_in[3];
    const float* pe_b   = (const float*)d_in[4];
    const float* pe_g   = (const float*)d_in[5];
    const float* pe_bt  = (const float*)d_in[6];
    const float* in_w   = (const float*)d_in[7];
    const float* in_b   = (const float*)d_in[8];
    const float* out_w  = (const float*)d_in[9];
    const float* out_b  = (const float*)d_in[10];
    float* out = (float*)d_out;

    const int n = in_sizes[0] / E;            // 2000 (multiple of 16)

    char* ws = (char*)d_ws;
    float* pos  = (float*)ws;  ws += (size_t)n * E * sizeof(float);
    float* cent = (float*)ws;  ws += (size_t)n * 4 * sizeof(float);
    float* qkv  = (float*)ws;  ws += (size_t)n * 3 * E * sizeof(float);
    float* ov   = (float*)ws;  ws += (size_t)n * E * sizeof(float);
    int*   nbr  = (int*)ws;    // n * KNBR ints

    prep_kernel<<<n, 32, 0, stream>>>(x, coords, prompt, pe_w, pe_b, pe_g, pe_bt, pos, cent, n);
    topk_kernel<<<n, 256, 0, stream>>>(cent, nbr, n);
    qkv_kernel<<<dim3(3 * E / 16, n / 16), 32, 0, stream>>>(pos, in_w, in_b, qkv);
    attn_kernel<<<n, 256, 0, stream>>>(qkv, nbr, ov);
    outproj_kernel<<<dim3(E / 16, n / 16), 32, 0, stream>>>(ov, out_w, out_b, out);
}
